// Bert_BiLstm_Crf_84662395339019
// MI455X (gfx1250) — compile-verified, matrix operand order checked by reading, not run on hardware
//
#include <hip/hip_runtime.h>
#include <hip/hip_bf16.h>

typedef unsigned short ushort_t;
typedef unsigned char  uchar_t;
typedef __attribute__((ext_vector_type(16))) __bf16 v16bf;
typedef __attribute__((ext_vector_type(8)))  float  v8f;

union Frag16 { v16bf v; uint4 q[2]; };

__device__ __forceinline__ ushort_t f2bf(float x) {
  unsigned int u = __float_as_uint(x);
  u += 0x7FFFu + ((u >> 16) & 1u);   // round-to-nearest-even
  return (ushort_t)(u >> 16);
}
__device__ __forceinline__ float bf2f(ushort_t h) {
  return __uint_as_float(((unsigned int)h) << 16);
}
__device__ __forceinline__ float sigmoidf_(float x) {
  return 1.0f / (1.0f + __expf(-x));
}

// ---------------------------------------------------------------------------
// f32 -> bf16 conversion (grid-stride)
// ---------------------------------------------------------------------------
__global__ void f32_to_bf16_k(const float* __restrict__ in,
                              ushort_t* __restrict__ out, int n) {
  for (int i = blockIdx.x * blockDim.x + threadIdx.x; i < n;
       i += gridDim.x * blockDim.x)
    out[i] = f2bf(in[i]);
}

// ---------------------------------------------------------------------------
// C[M,N](bf16) = A[M,K](bf16, row-major) x B[N,K](bf16, row-major)^T
// Block: 256 threads = 8 waves in a 4(M) x 2(N) grid; wave tile 32x64.
// K stepped by 32 with v_wmma_f32_16x16x32_bf16, f32 accumulate.
// ---------------------------------------------------------------------------
__global__ __launch_bounds__(256) void gemm_bf16_nt(
    const ushort_t* __restrict__ A, const ushort_t* __restrict__ B,
    ushort_t* __restrict__ C, int M, int N, int K) {
  const int lane = threadIdx.x & 31;
  const int w    = threadIdx.x >> 5;
  const int wm   = w & 3;
  const int wn   = w >> 2;
  const int m0   = blockIdx.x * 128 + wm * 32;
  const int n0   = blockIdx.y * 128 + wn * 64;
  const int rsel = lane & 15;
  const int koff = (lane >> 4) << 3;   // 16-bit A/B fragment K-split per lane half

  v8f acc[2][4];
#pragma unroll
  for (int i = 0; i < 2; ++i)
#pragma unroll
    for (int j = 0; j < 4; ++j)
      acc[i][j] = (v8f){0.f, 0.f, 0.f, 0.f, 0.f, 0.f, 0.f, 0.f};

  for (int k0 = 0; k0 < K; k0 += 32) {
    Frag16 a[2], b[4];
#pragma unroll
    for (int i = 0; i < 2; ++i) {
      const ushort_t* p = A + (size_t)(m0 + i * 16 + rsel) * K + k0 + koff;
      a[i].q[0] = *(const uint4*)p;
      a[i].q[1] = *(const uint4*)(p + 16);
    }
#pragma unroll
    for (int j = 0; j < 4; ++j) {
      const ushort_t* p = B + (size_t)(n0 + j * 16 + rsel) * K + k0 + koff;
      b[j].q[0] = *(const uint4*)p;
      b[j].q[1] = *(const uint4*)(p + 16);
    }
#pragma unroll
    for (int i = 0; i < 2; ++i)
#pragma unroll
      for (int j = 0; j < 4; ++j)
        acc[i][j] = __builtin_amdgcn_wmma_f32_16x16x32_bf16(
            false, a[i].v, false, b[j].v, (short)0, acc[i][j], false, false);
  }

  const int rhalf = (lane >> 4) << 3;
#pragma unroll
  for (int i = 0; i < 2; ++i)
#pragma unroll
    for (int j = 0; j < 4; ++j)
#pragma unroll
      for (int v = 0; v < 8; ++v) {
        const int row = m0 + i * 16 + v + rhalf;
        const int col = n0 + j * 16 + rsel;
        C[(size_t)row * N + col] = f2bf(acc[i][j][v]);
      }
}

// ---------------------------------------------------------------------------
// Persistent LSTM scan. grid = (4 batch tiles of 16, 2 directions).
// Per step: gates[16,1536] = h_prev[16,384] @ W_hh^T via WMMA (h in LDS, W_hh
// streamed from L2), then fused pre+bias+sigmoid/tanh cell update in LDS.
// pre[t+1] tile (16x1536 bf16 = 48KB) is double-buffered into LDS with
// GLOBAL_LOAD_ASYNC_TO_LDS_B128 (ASYNCcnt), hidden behind each step's compute.
// pre layout: [B*T, 1536] bf16 (row = b*512+t). hout: [B*T, 768] bf16.
// ---------------------------------------------------------------------------
// LDS: gates 96KB + c 24KB + bsum 6KB + hA 12.25KB + pre double buffer 96KB
#define SCAN_LDS_BYTES \
  ((16 * 1536 + 16 * 384 + 1536) * 4 + 16 * 392 * 2 + 2 * 16 * 1536 * 2)

__global__ __launch_bounds__(256) void lstm_scan(
    const ushort_t* __restrict__ preF, const ushort_t* __restrict__ preB,
    const ushort_t* __restrict__ whhF, const ushort_t* __restrict__ whhB,
    const float* __restrict__ bihF, const float* __restrict__ bhhF,
    const float* __restrict__ bihB, const float* __restrict__ bhhB,
    ushort_t* __restrict__ hout) {
  extern __shared__ char smem_raw[];
  float*    gates = (float*)smem_raw;          // [16][1536] f32
  float*    cS    = gates + 16 * 1536;         // [16][384]  f32 cell state
  float*    bsum  = cS + 16 * 384;             // [1536] b_ih + b_hh
  ushort_t* hA    = (ushort_t*)(bsum + 1536);  // [16][392] bf16 h (A-frag layout)
  ushort_t* preL  = hA + 16 * 392;             // [2][16][1536] bf16 pre tiles

  const int mb  = blockIdx.x;   // batch tile (16 rows)
  const int dir = blockIdx.y;   // 0=forward, 1=backward
  const ushort_t* pre = dir ? preB : preF;
  const ushort_t* whh = dir ? whhB : whhF;
  const float*    bih = dir ? bihB : bihF;
  const float*    bhh = dir ? bhhB : bhhF;
  const int colOff = dir ? 384 : 0;

  const int tid   = threadIdx.x;
  const int lane  = tid & 31;
  const int wid   = tid >> 5;
  const int rsel  = lane & 15;
  const int koff  = (lane >> 4) << 3;
  const int rhalf = (lane >> 4) << 3;

  for (int i = tid; i < 1536; i += 256) bsum[i] = bih[i] + bhh[i];
  for (int i = tid; i < 16 * 384; i += 256) cS[i] = 0.f;
  for (int i = tid; i < 16 * 392; i += 256) hA[i] = 0;

  // Async prefetch of the 16x1536 bf16 pre tile for time t into LDS buffer
  // `buf`. 3072 16-byte chunks across 256 threads = 12 issues/thread; each
  // enabled lane moves 16B (b128) memory->LDS, tracked by ASYNCcnt.
  auto prefetch_pre = [&](int t, int buf) {
    const unsigned ldsBase =
        (unsigned)(uintptr_t)(preL + buf * (16 * 1536));
#pragma unroll
    for (int w = 0; w < 12; ++w) {
      const int idx = tid + (w << 8);
      const int row = idx / 192;        // 192 16B chunks per 3072B row
      const int ch  = idx - row * 192;
      const unsigned lds = ldsBase + (unsigned)(row * 3072 + ch * 16);
      const ushort_t* gp =
          pre + ((size_t)(mb * 16 + row) * 512 + (size_t)t) * 1536 + ch * 8;
      const unsigned long long ga = (unsigned long long)(uintptr_t)gp;
      asm volatile("global_load_async_to_lds_b128 %0, %1, off"
                   :: "v"(lds), "v"(ga)
                   : "memory");
    }
  };

  prefetch_pre(dir ? 511 : 0, 0);
  asm volatile("s_wait_asynccnt 0x0" ::: "memory");
  __syncthreads();

  for (int s = 0; s < 512; ++s) {
    const int t = dir ? (511 - s) : s;

    // Kick off next step's pre tile while this step computes.
    if (s < 511) prefetch_pre(dir ? (510 - s) : (s + 1), (s + 1) & 1);

    // A fragments (h_prev, 16x384) from LDS, cached across the 12 N-tiles.
    Frag16 aF[12];
#pragma unroll
    for (int kk = 0; kk < 12; ++kk) {
      const ushort_t* p = hA + rsel * 392 + (kk << 5) + koff;
      aF[kk].q[0] = *(const uint4*)p;
      aF[kk].q[1] = *(const uint4*)(p + 16);
    }

    // 96 N-tiles of 16 across 8 waves -> 12 tiles/wave, 12 WMMAs each.
    for (int i = 0; i < 12; ++i) {
      const int n0 = (wid + (i << 3)) << 4;
      v8f acc = (v8f){0.f, 0.f, 0.f, 0.f, 0.f, 0.f, 0.f, 0.f};
#pragma unroll
      for (int kk = 0; kk < 12; ++kk) {
        Frag16 b;
        const ushort_t* p = whh + (size_t)(n0 + rsel) * 384 + (kk << 5) + koff;
        b.q[0] = *(const uint4*)p;
        b.q[1] = *(const uint4*)(p + 16);
        acc = __builtin_amdgcn_wmma_f32_16x16x32_bf16(
            false, aF[kk].v, false, b.v, (short)0, acc, false, false);
      }
#pragma unroll
      for (int v = 0; v < 8; ++v)
        gates[(v + rhalf) * 1536 + n0 + rsel] = acc[v];
    }
    __syncthreads();

    // Fused cell update: i,f,g,o at column offsets 0/384/768/1152.
    // pre comes from the LDS tile prefetched during the previous step.
    const ushort_t* plBase = preL + (s & 1) * (16 * 1536);
    for (int u = tid; u < 16 * 384; u += 256) {
      const int r = u / 384;
      const int j = u - r * 384;
      const ushort_t* pl = plBase + r * 1536;
      const float gi = gates[r * 1536 + j       ] + bf2f(pl[j       ]) + bsum[j       ];
      const float gf = gates[r * 1536 + j +  384] + bf2f(pl[j +  384]) + bsum[j +  384];
      const float gg = gates[r * 1536 + j +  768] + bf2f(pl[j +  768]) + bsum[j +  768];
      const float go = gates[r * 1536 + j + 1152] + bf2f(pl[j + 1152]) + bsum[j + 1152];
      const float iv = sigmoidf_(gi);
      const float fv = sigmoidf_(gf);
      const float gv = tanhf(gg);
      const float ov = sigmoidf_(go);
      const float c  = fv * cS[u] + iv * gv;
      cS[u] = c;
      const float h = ov * tanhf(c);
      const ushort_t hb = f2bf(h);
      hA[r * 392 + j] = hb;
      hout[((size_t)(mb * 16 + r) * 512 + (size_t)t) * 768 + colOff + j] = hb;
    }
    // Next step reads the freshly prefetched pre tile: drain ASYNCcnt, then
    // rendezvous so every wave sees the new tile and this step's h/c.
    asm volatile("s_wait_asynccnt 0x0" ::: "memory");
    __syncthreads();
  }
}

// ---------------------------------------------------------------------------
// feats[B*T, 30] = h2[B*T, 768](bf16) @ fc_w^T(bf16) + fc_b.
// One WG = 8 rows; fc_w and rows staged in LDS; lane<30 does a dot each.
// ---------------------------------------------------------------------------
__global__ __launch_bounds__(256) void fc_kernel(
    const ushort_t* __restrict__ h2, const ushort_t* __restrict__ fcw,
    const float* __restrict__ fcb, float* __restrict__ feats) {
  __shared__ ushort_t wl[30 * 768];
  __shared__ ushort_t rowl[8 * 768];
  const int tid  = threadIdx.x;
  const int lane = tid & 31;
  const int wid  = tid >> 5;
  const size_t rbase = (size_t)blockIdx.x * 8;

  for (int i = tid; i < 30 * 768 / 2; i += 256)
    ((unsigned int*)wl)[i] = ((const unsigned int*)fcw)[i];
  for (int i = tid; i < 8 * 768 / 2; i += 256)
    ((unsigned int*)rowl)[i] = ((const unsigned int*)h2)[rbase * 384 + i];
  __syncthreads();

  if (lane < 30) {
    float acc = 0.f;
    const ushort_t* rp = rowl + wid * 768;
    const ushort_t* wp = wl + lane * 768;
#pragma unroll 8
    for (int k = 0; k < 768; ++k) acc += bf2f(rp[k]) * bf2f(wp[k]);
    feats[(rbase + wid) * 30 + lane] = acc + fcb[lane];
  }
}

// ---------------------------------------------------------------------------
// Viterbi: one wave per batch element. Faithful to reference: ld starts at
// -10000 everywhere; feats[:,0] never added; psi for steps t=1..511.
// ---------------------------------------------------------------------------
__global__ __launch_bounds__(32) void viterbi_kernel(
    const float* __restrict__ feats, const float* __restrict__ trans,
    uchar_t* __restrict__ psi, float* __restrict__ score,
    int* __restrict__ path) {
  __shared__ float tr[900];
  __shared__ float ld[32];
  __shared__ float ldn[32];
  const int b = blockIdx.x;
  const int j = threadIdx.x;
  for (int i = j; i < 900; i += 32) tr[i] = trans[i];
  ld[j] = -10000.0f;
  __syncthreads();

  for (int t = 1; t < 512; ++t) {
    if (j < 30) {
      float best = -3.4e38f;
      int   barg = 0;
      for (int p = 0; p < 30; ++p) {
        const float v = tr[j * 30 + p] + ld[p];
        if (v > best) { best = v; barg = p; }
      }
      ldn[j] = best + feats[((size_t)b * 512 + t) * 30 + j];
      psi[((size_t)b * 511 + (t - 1)) * 30 + j] = (uchar_t)barg;
    }
    __syncthreads();
    if (j < 30) ld[j] = ldn[j];
    __syncthreads();
  }

  if (j == 0) {
    float best = ld[0];
    int   last = 0;
    for (int p = 1; p < 30; ++p)
      if (ld[p] > best) { best = ld[p]; last = p; }
    score[b] = best;
    int cur = last;
    path[(size_t)b * 512 + 511] = cur;
    for (int t = 510; t >= 0; --t) {
      cur = psi[((size_t)b * 511 + t) * 30 + cur];
      path[(size_t)b * 512 + t] = cur;
    }
  }
}

// ---------------------------------------------------------------------------
extern "C" void kernel_launch(void* const* d_in, const int* in_sizes, int n_in,
                              void* d_out, int out_size, void* d_ws,
                              size_t ws_size, hipStream_t stream) {
  (void)in_sizes; (void)n_in; (void)out_size; (void)ws_size;
  const float* embeds = (const float*)d_in[0];
  const float* trans  = (const float*)d_in[1];
  const float* fc_w   = (const float*)d_in[2];
  const float* fc_b   = (const float*)d_in[3];
  // order: [0]=l0f, [1]=l0b, [2]=l1f, [3]=l1b
  const float* w_ih[4] = {(const float*)d_in[4],  (const float*)d_in[8],
                          (const float*)d_in[12], (const float*)d_in[16]};
  const float* w_hh[4] = {(const float*)d_in[5],  (const float*)d_in[9],
                          (const float*)d_in[13], (const float*)d_in[17]};
  const float* b_ih[4] = {(const float*)d_in[6],  (const float*)d_in[10],
                          (const float*)d_in[14], (const float*)d_in[18]};
  const float* b_hh[4] = {(const float*)d_in[7],  (const float*)d_in[11],
                          (const float*)d_in[15], (const float*)d_in[19]};

  char* ws = (char*)d_ws;
  size_t off = 0;
  auto alloc = [&](size_t bytes) {
    char* p = ws + off;
    off += (bytes + 255) & ~(size_t)255;
    return p;
  };
  ushort_t* Xb   = (ushort_t*)alloc((size_t)32768 * 768 * 2);   // embeds bf16; reused as h2
  ushort_t* preF = (ushort_t*)alloc((size_t)32768 * 1536 * 2);  // per-layer fwd pre
  ushort_t* preB = (ushort_t*)alloc((size_t)32768 * 1536 * 2);  // per-layer bwd pre
  ushort_t* h1   = (ushort_t*)alloc((size_t)32768 * 768 * 2);   // layer-0 output
  ushort_t* wihb[4];
  for (int i = 0; i < 4; ++i) wihb[i] = (ushort_t*)alloc((size_t)1536 * 768 * 2);
  ushort_t* whhb[4];
  for (int i = 0; i < 4; ++i) whhb[i] = (ushort_t*)alloc((size_t)1536 * 384 * 2);
  ushort_t* fcwb  = (ushort_t*)alloc((size_t)30 * 768 * 2);
  float*    feats = (float*)alloc((size_t)32768 * 30 * 4);
  uchar_t*  psi   = (uchar_t*)alloc((size_t)64 * 511 * 30);

  float* score_out = (float*)d_out;
  int*   path_out  = (int*)(score_out + 64);

  hipFuncSetAttribute(reinterpret_cast<const void*>(lstm_scan),
                      hipFuncAttributeMaxDynamicSharedMemorySize,
                      SCAN_LDS_BYTES);

  // Precision staging: everything matmul-bound goes bf16.
  f32_to_bf16_k<<<4096, 256, 0, stream>>>(embeds, Xb, 32768 * 768);
  for (int i = 0; i < 4; ++i)
    f32_to_bf16_k<<<1024, 256, 0, stream>>>(w_ih[i], wihb[i], 1536 * 768);
  for (int i = 0; i < 4; ++i)
    f32_to_bf16_k<<<512, 256, 0, stream>>>(w_hh[i], whhb[i], 1536 * 384);
  f32_to_bf16_k<<<32, 256, 0, stream>>>(fc_w, fcwb, 30 * 768);

  const dim3 ggrid(32768 / 128, 1536 / 128);

  // Layer 0: input projections (big WMMA GEMMs), then batch-parallel scans.
  gemm_bf16_nt<<<ggrid, 256, 0, stream>>>(Xb, wihb[0], preF, 32768, 1536, 768);
  gemm_bf16_nt<<<ggrid, 256, 0, stream>>>(Xb, wihb[1], preB, 32768, 1536, 768);
  lstm_scan<<<dim3(4, 2), 256, SCAN_LDS_BYTES, stream>>>(
      preF, preB, whhb[0], whhb[1], b_ih[0], b_hh[0], b_ih[1], b_hh[1], h1);

  // Layer 1 (pre buffers reused; h2 written into the embeds buffer).
  gemm_bf16_nt<<<ggrid, 256, 0, stream>>>(h1, wihb[2], preF, 32768, 1536, 768);
  gemm_bf16_nt<<<ggrid, 256, 0, stream>>>(h1, wihb[3], preB, 32768, 1536, 768);
  lstm_scan<<<dim3(4, 2), 256, SCAN_LDS_BYTES, stream>>>(
      preF, preB, whhb[2], whhb[3], b_ih[2], b_hh[2], b_ih[3], b_hh[3], Xb);

  // FC + CRF Viterbi decode.
  fc_kernel<<<4096, 256, 0, stream>>>(Xb, fcwb, fc_b, feats);
  viterbi_kernel<<<64, 32, 0, stream>>>(feats, trans, psi, score_out, path_out);
}
